// BiRNNModel_20779051778085
// MI455X (gfx1250) — matrix operand order repaired and checked by reading, not compiled, not run on hardware
//
#include <hip/hip_runtime.h>
#include <stdint.h>
#include <stddef.h>

// ---------------- CDNA5 WMMA / TDM types ----------------
typedef __attribute__((ext_vector_type(8)))  float    v8f;
typedef __attribute__((ext_vector_type(8)))  __bf16   v8bf;
typedef __attribute__((ext_vector_type(16))) __bf16   v16bf;
typedef __attribute__((ext_vector_type(4)))  uint32_t u32x4;
typedef __attribute__((ext_vector_type(4)))  int      i32x4;
typedef __attribute__((ext_vector_type(8)))  int      i32x8;

#if defined(__has_builtin)
#  if __has_builtin(__builtin_amdgcn_tensor_load_to_lds) && __has_builtin(__builtin_amdgcn_s_wait_tensorcnt)
#    define USE_TDM 1
#  else
#    define USE_TDM 0
#  endif
#else
#  define USE_TDM 0
#endif

#define BSZ   32
#define TSZ   512
#define HSZ   1024
#define G4H   4096
#define ROWS  (BSZ * TSZ)          // 16384
#define LDS_ROW 1040               // padded LDS row stride in bf16 elems (2080 B)

// ---------------- helpers ----------------
__device__ __forceinline__ uint16_t f32_to_bf16_rne(float f) {
  union { float f; uint32_t u; } v; v.f = f;
  uint32_t u = v.u;
  u += 0x7fffu + ((u >> 16) & 1u);   // round-to-nearest-even
  return (uint16_t)(u >> 16);
}
__device__ __forceinline__ float bf16_to_f32(uint16_t h) {
  union { uint32_t u; float f; } v; v.u = ((uint32_t)h) << 16;
  return v.f;
}
// Build a 16-element bf16 fragment from two 16-byte chunks.
__device__ __forceinline__ v16bf load_frag16(const uint16_t* p0, const uint16_t* p1) {
  v8bf lo = *reinterpret_cast<const v8bf*>(p0);
  v8bf hi = *reinterpret_cast<const v8bf*>(p1);
  v16bf r;
#pragma unroll
  for (int i = 0; i < 8; ++i) { r[i] = lo[i]; r[i + 8] = hi[i]; }
  return r;
}
__device__ __forceinline__ float sigmoidf_fast(float x) {
  return 1.0f / (1.0f + __expf(-x));
}

// Device-scope software barrier across nwg co-resident workgroups.
__device__ __forceinline__ void grid_barrier(int* cnt, int* gen, int nwg) {
  __threadfence();
  __syncthreads();
  if (threadIdx.x == 0) {
    int g0 = __hip_atomic_load(gen, __ATOMIC_RELAXED, __HIP_MEMORY_SCOPE_AGENT);
    int prev = __hip_atomic_fetch_add(cnt, 1, __ATOMIC_ACQ_REL, __HIP_MEMORY_SCOPE_AGENT);
    if (prev == nwg - 1) {
      __hip_atomic_store(cnt, 0, __ATOMIC_RELAXED, __HIP_MEMORY_SCOPE_AGENT);
      __hip_atomic_fetch_add(gen, 1, __ATOMIC_RELEASE, __HIP_MEMORY_SCOPE_AGENT);
    } else {
      while (__hip_atomic_load(gen, __ATOMIC_ACQUIRE, __HIP_MEMORY_SCOPE_AGENT) == g0)
        __builtin_amdgcn_s_sleep(1);
    }
  }
  __syncthreads();
}

// ---------------- f32 -> bf16 conversion ----------------
__global__ __launch_bounds__(256) void cvt_f32_bf16(const float* __restrict__ s,
                                                    uint16_t* __restrict__ d, int n) {
  int i = blockIdx.x * 256 + threadIdx.x;
  if (i < n) d[i] = f32_to_bf16_rne(s[i]);
}

// ---------------- input-projection GEMM ----------------
// C[r,n] = sum_k A[r,k] * W[n,k]    A:[16384,1024] bf16, W:[4096,1024] bf16
// C:[16384,4096] stored bf16. Block tile 32M x 256N, 8 waves, 4 N-tiles/wave.
// A panel (32x1024 bf16 = 64 KB) is DMA'd into LDS once per block by the
// Tensor Data Mover (with LDS padding for bank spread), then A fragments come
// from ds_load_b128 while B streams from the L2-resident weights.
__global__ __launch_bounds__(256) void xproj_gemm(const uint16_t* __restrict__ A,
                                                  const uint16_t* __restrict__ W,
                                                  uint16_t* __restrict__ C) {
#if USE_TDM
  __shared__ uint16_t lds_a[32 * LDS_ROW];   // 65 KB padded A panel
#endif
  const int lane = threadIdx.x & 31;
  const int wv   = threadIdx.x >> 5;
  const int l15  = lane & 15;
  const int m0   = blockIdx.x * 32 + (wv & 1) * 16;     // row tile base
  const int n0   = blockIdx.y * 256 + (wv >> 1) * 64;   // 4 n-tiles of 16
  const int akb  = (lane >> 4) << 3;                    // A frag K sub-base (0/8)
  const int bk0  = (lane >> 4) << 4;                    // unused (0/16); B frags use akb

#if USE_TDM
  if (wv == 0) {
    // ---- D# descriptor: 2D tile, tile_dim0=1024, tile_dim1=32, 2-byte elems.
    // pad_enable=1, pad_interval=7 (256 DWORDs = 1024 B), pad_amount=3 (4 DWORDs
    // = 16 B) -> LDS row stride 2048+2*16 = 2080 B, fragments stay 16B aligned.
    uint64_t ga = (uint64_t)(uintptr_t)(A + (size_t)blockIdx.x * 32 * HSZ);
    uint32_t la = (uint32_t)(uintptr_t)&lds_a[0];
    u32x4 g0;
    g0[0] = 1u;                                            // count=1, user D#
    g0[1] = la;                                            // lds_addr
    g0[2] = (uint32_t)ga;                                  // global_addr[31:0]
    g0[3] = (uint32_t)((ga >> 32) & 0x1FFFFFFu) | (2u << 30); // addr[56:32] | type=2
    i32x8 g1;
    g1[0] = (1 << 16) | (1 << 20) | (7 << 22) | (3 << 25); // data_size=2B, pad cfg
    g1[1] = (int)(1024u << 16);                            // tensor_dim0 = 1024
    g1[2] = (int)(16384u << 16);                           // tensor_dim1 = 16384
    g1[3] = (int)(1024u << 16);                            // tile_dim0 = 1024
    g1[4] = 32;                                            // tile_dim1 = 32
    g1[5] = 1024;                                          // tensor_dim0_stride
    g1[6] = 0;
    g1[7] = 0;
    i32x4 z4 = {0, 0, 0, 0};
#if defined(__clang_major__) && __clang_major__ >= 23
    i32x8 z8 = {0, 0, 0, 0, 0, 0, 0, 0};
    __builtin_amdgcn_tensor_load_to_lds(g0, g1, z4, z4, z8, 0);
#else
    __builtin_amdgcn_tensor_load_to_lds(g0, g1, z4, z4, 0);
#endif
    __builtin_amdgcn_s_wait_tensorcnt(0);
  }
  __syncthreads();
  const uint16_t* arow = &lds_a[(size_t)((wv & 1) * 16 + l15) * LDS_ROW];
#else
  const uint16_t* arp = A + (size_t)(m0 + l15) * HSZ;
#endif
  const uint16_t* w0 = W + (size_t)(n0 +  0 + l15) * HSZ;
  const uint16_t* w1 = W + (size_t)(n0 + 16 + l15) * HSZ;
  const uint16_t* w2 = W + (size_t)(n0 + 32 + l15) * HSZ;
  const uint16_t* w3 = W + (size_t)(n0 + 48 + l15) * HSZ;

  v8f acc0 = {}, acc1 = {}, acc2 = {}, acc3 = {};
#pragma unroll 4
  for (int kk = 0; kk < HSZ; kk += 32) {
#if USE_TDM
    // padded element offset: +8 elems (16 B) after each 512 elems (1024 B)
    int e0 = kk + akb;
    int e1 = e0 + 16;
    int p0 = e0 + ((e0 >= 512) ? 8 : 0);
    int p1 = e1 + ((e1 >= 512) ? 8 : 0);
    v16bf a = load_frag16(arow + p0, arow + p1);
#else
    v16bf a = load_frag16(arp + kk + akb, arp + kk + akb + 16);
#endif
    v16bf b0 = load_frag16(w0 + kk + akb, w0 + kk + akb + 16);
    v16bf b1 = load_frag16(w1 + kk + akb, w1 + kk + akb + 16);
    v16bf b2 = load_frag16(w2 + kk + akb, w2 + kk + akb + 16);
    v16bf b3 = load_frag16(w3 + kk + akb, w3 + kk + akb + 16);
    acc0 = __builtin_amdgcn_wmma_f32_16x16x32_bf16(false, a, false, b0, (short)0, acc0, false, false);
    acc1 = __builtin_amdgcn_wmma_f32_16x16x32_bf16(false, a, false, b1, (short)0, acc1, false, false);
    acc2 = __builtin_amdgcn_wmma_f32_16x16x32_bf16(false, a, false, b2, (short)0, acc2, false, false);
    acc3 = __builtin_amdgcn_wmma_f32_16x16x32_bf16(false, a, false, b3, (short)0, acc3, false, false);
  }
  // C/D layout: VGPR r -> row m0 + r + (lane>=16 ? 8 : 0), col = n-tile base + (lane&15)
  const int rb = m0 + ((lane >> 4) << 3);
#pragma unroll
  for (int r = 0; r < 8; ++r) {
    size_t row = (size_t)(rb + r) * G4H;
    C[row + n0 +  0 + l15] = f32_to_bf16_rne(acc0[r]);
    C[row + n0 + 16 + l15] = f32_to_bf16_rne(acc1[r]);
    C[row + n0 + 32 + l15] = f32_to_bf16_rne(acc2[r]);
    C[row + n0 + 48 + l15] = f32_to_bf16_rne(acc3[r]);
  }
}

// ---------------- persistent bidirectional LSTM recurrence ----------------
// Grid: 128 blocks x 256 threads. blocks [0,64) = forward, [64,128) = backward.
// Each WG owns 16 hidden units (cell state in registers). Per step each wave
// computes one 16x16 gate tile of h @ Whh^T via 32 chained WMMAs; h is
// double-buffered in global bf16 and exchanged through a per-direction
// software grid barrier.
__global__ __launch_bounds__(256) void lstm_recurrence(
    const uint16_t* __restrict__ xprojF, const uint16_t* __restrict__ xprojB,
    const uint16_t* __restrict__ whhF,   const uint16_t* __restrict__ whhB,
    const float* __restrict__ biasF,     const float* __restrict__ biasB,
    uint16_t* __restrict__ hF0, uint16_t* __restrict__ hF1,
    uint16_t* __restrict__ hB0, uint16_t* __restrict__ hB1,
    uint16_t* __restrict__ seqOutF, uint16_t* __restrict__ seqOutB,  // bf16 [16384,1024] or null
    float* __restrict__ outCat,                                      // [16384,2048] or null
    float* __restrict__ outFwd, float* __restrict__ outBwd,          // [16384,1024] or null
    int* __restrict__ barriers)                                      // {cntF,genF,cntB,genB}
{
  __shared__ float gates[BSZ * 64];   // 8 KB: 32 rows x (4 gates x 16 units)

  const int dir  = blockIdx.x >> 6;
  const int wg   = blockIdx.x & 63;
  const int j0   = wg * 16;
  const int lane = threadIdx.x & 31;
  const int l15  = lane & 15;
  const int wv   = threadIdx.x >> 5;
  const int mtl  = wv & 1;            // which 16-row half of B=32
  const int g    = wv >> 1;           // gate block 0..3

  const uint16_t* xproj = dir ? xprojB : xprojF;
  const uint16_t* whh   = dir ? whhB   : whhF;
  const float*    bias  = dir ? biasB  : biasF;
  uint16_t* hbuf[2];
  hbuf[0] = dir ? hB0 : hF0;
  hbuf[1] = dir ? hB1 : hF1;
  uint16_t* seqOut = dir ? seqOutB : seqOutF;
  int* cnt = barriers + dir * 2;
  int* gen = cnt + 1;

  // Per-thread cell state: 2 (b, jj) pairs (32 b x 16 jj = 512 per WG)
  const int idx0 = threadIdx.x * 2;
  float cst[2] = {0.f, 0.f};

  // Zero this WG's slice of h ping buffer (initial hidden state)
#pragma unroll
  for (int p = 0; p < 2; ++p) {
    int idx = idx0 + p;
    hbuf[0][(idx >> 4) * HSZ + j0 + (idx & 15)] = 0;
  }
  grid_barrier(cnt, gen, 64);

  // Constant B-operand row pointer: gates column n = g*1024 + j0 + (lane&15)
  const uint16_t* wr = whh + (size_t)(g * HSZ + j0 + l15) * HSZ;
  const int akb = (lane >> 4) << 3;   // 0 / 8
  const int bk0 = (lane >> 4) << 4;   // 0 / 16
  const int arow = mtl * 16 + l15;

  int pp = 0;                          // ping-pong selector
  for (int s = 0; s < TSZ; ++s) {
    const int t = dir ? (TSZ - 1 - s) : s;
    const uint16_t* hprev = hbuf[pp];
    uint16_t*       hnext = hbuf[pp ^ 1];

    // ---- recurrent GEMM tile: [16 x 16] over K=1024 ----
    const uint16_t* arp = hprev + (size_t)arow * HSZ;
    v8f acc = {};
#pragma unroll 8
    for (int kk = 0; kk < HSZ; kk += 32) {
      v16bf a = load_frag16(arp + kk + akb, arp + kk + akb + 16);
      v16bf b = load_frag16(wr + kk + akb, wr + kk + akb + 16);
      acc = __builtin_amdgcn_wmma_f32_16x16x32_bf16(false, a, false, b, (short)0, acc, false, false);
    }
    // spill tile to LDS: row = mtl*16 + r + (lane>=16?8:0), local col = g*16 + (lane&15)
    {
      const int rb = mtl * 16 + ((lane >> 4) << 3);
      const int lc = g * 16 + l15;
#pragma unroll
      for (int r = 0; r < 8; ++r) gates[(rb + r) * 64 + lc] = acc[r];
    }
    __syncthreads();

    // ---- elementwise gate math ----
#pragma unroll
    for (int p = 0; p < 2; ++p) {
      int idx = idx0 + p;
      int b  = idx >> 4;
      int jj = idx & 15;
      size_t row = (size_t)b * TSZ + t;
      const uint16_t* xp = xproj + row * G4H + j0 + jj;
      int col = j0 + jj;
      float gi = gates[b * 64 +      jj] + bf16_to_f32(xp[0])    + bias[       col];
      float gf = gates[b * 64 + 16 + jj] + bf16_to_f32(xp[1024]) + bias[1024 + col];
      float gg = gates[b * 64 + 32 + jj] + bf16_to_f32(xp[2048]) + bias[2048 + col];
      float go = gates[b * 64 + 48 + jj] + bf16_to_f32(xp[3072]) + bias[3072 + col];
      float cc = sigmoidf_fast(gf) * cst[p] + sigmoidf_fast(gi) * tanhf(gg);
      cst[p] = cc;
      float h = sigmoidf_fast(go) * tanhf(cc);
      hnext[b * HSZ + col] = f32_to_bf16_rne(h);
      if (seqOut) seqOut[row * HSZ + col] = f32_to_bf16_rne(h);
      if (outCat) {
        outCat[row * 2048 + dir * HSZ + col] = h;
        (dir ? outBwd : outFwd)[row * HSZ + col] = h;
      }
    }
    grid_barrier(cnt, gen, 64);   // publishes hnext to all WGs of this direction
    pp ^= 1;
  }
}

// ---------------- host driver ----------------
extern "C" void kernel_launch(void* const* d_in, const int* in_sizes, int n_in,
                              void* d_out, int out_size, void* d_ws, size_t ws_size,
                              hipStream_t stream) {
  (void)in_sizes; (void)n_in; (void)out_size; (void)ws_size;
  const float* x    = (const float*)d_in[0];
  const float* WihF = (const float*)d_in[1];
  const float* WhhF = (const float*)d_in[2];
  const float* bF   = (const float*)d_in[3];
  const float* WihB = (const float*)d_in[4];
  const float* WhhB = (const float*)d_in[5];
  const float* bB   = (const float*)d_in[6];

  char* ws = (char*)d_ws;
  const size_t MB = 1u << 20;
  const size_t WELEM = (size_t)G4H * HSZ;          // 4,194,304 elems per weight matrix
  uint16_t* wih   = (uint16_t*)(ws);               // [dir*2+layer] x 4M bf16 (32 MB)
  uint16_t* whh   = (uint16_t*)(ws +  32 * MB);    // 32 MB
  uint16_t* xbf   = (uint16_t*)(ws +  64 * MB);    // x  bf16 (32 MB)
  uint16_t* seq1f = (uint16_t*)(ws +  96 * MB);    // layer-0 fwd output, bf16 (32 MB)
  uint16_t* seq1b = (uint16_t*)(ws + 128 * MB);    // layer-0 bwd output, bf16 (32 MB)
  uint16_t* xprF  = (uint16_t*)(ws + 160 * MB);    // Xproj fwd, bf16 (128 MB)
  uint16_t* xprB  = (uint16_t*)(ws + 288 * MB);    // Xproj bwd, bf16 (128 MB)
  uint16_t* hF0   = (uint16_t*)(ws + 416 * MB);    // h ping/pong, 4 x 64 KB
  uint16_t* hF1   = hF0 + BSZ * HSZ;
  uint16_t* hB0   = hF1 + BSZ * HSZ;
  uint16_t* hB1   = hB0 + BSZ * HSZ;
  int*      bar   = (int*)(ws + 416 * MB + 512 * 1024);

  hipMemsetAsync(bar, 0, 64, stream);

  // ---- convert weights and input to bf16 ----
  const int WN = (int)(2 * WELEM);                 // 8M elems per direction (2 layers)
  cvt_f32_bf16<<<(WN + 255) / 256, 256, 0, stream>>>(WihF, wih, WN);
  cvt_f32_bf16<<<(WN + 255) / 256, 256, 0, stream>>>(WihB, wih + 2 * WELEM, WN);
  cvt_f32_bf16<<<(WN + 255) / 256, 256, 0, stream>>>(WhhF, whh, WN);
  cvt_f32_bf16<<<(WN + 255) / 256, 256, 0, stream>>>(WhhB, whh + 2 * WELEM, WN);
  const int XN = ROWS * HSZ;                       // 16M
  cvt_f32_bf16<<<(XN + 255) / 256, 256, 0, stream>>>(x, xbf, XN);

  dim3 ggrid(ROWS / 32, G4H / 256);                // 512 x 16

  // ---- layer 0 ----
  xproj_gemm<<<ggrid, 256, 0, stream>>>(xbf, wih + 0 * WELEM, xprF);
  xproj_gemm<<<ggrid, 256, 0, stream>>>(xbf, wih + 2 * WELEM, xprB);
  lstm_recurrence<<<128, 256, 0, stream>>>(
      xprF, xprB, whh + 0 * WELEM, whh + 2 * WELEM, bF, bB,
      hF0, hF1, hB0, hB1, seq1f, seq1b,
      nullptr, nullptr, nullptr, bar);

  // ---- layer 1 ----
  xproj_gemm<<<ggrid, 256, 0, stream>>>(seq1f, wih + 1 * WELEM, xprF);
  xproj_gemm<<<ggrid, 256, 0, stream>>>(seq1b, wih + 3 * WELEM, xprB);
  float* outCat = (float*)d_out;                               // [16384, 2048]
  float* outFwd = outCat + (size_t)ROWS * 2048;                // [16384, 1024]
  float* outBwd = outFwd + (size_t)ROWS * HSZ;                 // [16384, 1024]
  lstm_recurrence<<<128, 256, 0, stream>>>(
      xprF, xprB, whh + 1 * WELEM, whh + 3 * WELEM, bF + G4H, bB + G4H,
      hF0, hF1, hB0, hB1, nullptr, nullptr,
      outCat, outFwd, outBwd, bar);
}
